// GroupedVectorSA_38800734552580
// MI455X (gfx1250) — compile-verified, matrix operand order checked by reading, not run on hardware
//
#include <hip/hip_runtime.h>

// ---------------------------------------------------------------------------
// GroupedVectorSA for MI455X (gfx1250, wave32, WMMA).
// K0: BN-fold + bf16 conversion + pair-major weight packing.
// K1: fused q/k/v GEMM epilogue via v_wmma_f32_16x16x32_bf16.
// K2: per-point fused gather (GLOBAL_LOAD_ASYNC_TO_LDS, ASYNCcnt) ->
//     pos-MLPs (WMMA) -> relation -> attention weights (WMMA) -> softmax ->
//     grouped weighted sum, all in LDS.
// ---------------------------------------------------------------------------

#define DEVFN __device__ __forceinline__

typedef __attribute__((ext_vector_type(16))) __bf16 v16bf;
typedef __attribute__((ext_vector_type(8)))  float  v8f;

union FragU { unsigned u[8]; v16bf v; };

static constexpr int Bn = 2, Nn = 4096, Sn = 16, Cn = 256, Gn = 8;
static constexpr int ROWS = Bn * Nn;          // 8192 point rows

// ---- workspace layout (bytes) ----
static constexpr size_t OFF_Q      = 0;
static constexpr size_t OFF_K      = OFF_Q      + (size_t)ROWS * Cn * 4;
static constexpr size_t OFF_V      = OFF_K      + (size_t)ROWS * Cn * 4;
static constexpr size_t OFF_FEATSB = OFF_V      + (size_t)ROWS * Cn * 4;   // bf16 feats
static constexpr size_t OFF_WQ     = OFF_FEATSB + (size_t)ROWS * Cn * 2;   // pair-major bf16
static constexpr size_t OFF_WK     = OFF_WQ     + (size_t)Cn * Cn * 2;
static constexpr size_t OFF_WV     = OFF_WK     + (size_t)Cn * Cn * 2;
static constexpr size_t OFF_PMW2   = OFF_WV     + (size_t)Cn * Cn * 2;
static constexpr size_t OFF_PBW2   = OFF_PMW2   + (size_t)Cn * Cn * 2;
static constexpr size_t OFF_WEW1   = OFF_PBW2   + (size_t)Cn * Cn * 2;     // 256x16 padded
static constexpr size_t OFF_FOLD   = OFF_WEW1   + (size_t)Cn * 16 * 2;
// fold region (f32): s_q[256] t_q[256] s_k t_k s_pm t_pm s_pb t_pb s_we[8] t_we[8]

DEVFN unsigned short f2bf(float x) {
  union { float f; unsigned u; } c; c.f = x;
  unsigned r = c.u + 0x7FFFu + ((c.u >> 16) & 1u);   // round-to-nearest-even
  return (unsigned short)(r >> 16);
}

// A-fragment (16x32 bf16, MxK) from a row-major bf16 matrix viewed as dwords
// (2 consecutive K per dword).  ISA 7.12.2 layout:
//   lanes 0-15  : M=lane,    v0..v3 = K kb+0..7,  v4..v7 = K kb+16..23
//   lanes 16-31 : M=lane-16, v0..v3 = K kb+8..15, v4..v7 = K kb+24..31
DEVFN v16bf load_a_dw(const unsigned* base, int row, int strideDw, int kb, int lane) {
  FragU f;
  const unsigned* p = base + row * strideDw + ((kb + ((lane >> 4) << 3)) >> 1);
#pragma unroll
  for (int j = 0; j < 4; ++j) { f.u[j] = p[j]; f.u[4 + j] = p[8 + j]; }
  return f.v;
}

// B-fragment (32x16 bf16, KxN) from pair-major packed weights:
// dword d = (k>>1)*Npad + n holds {w[k][n], w[k+1][n]}.
DEVFN v16bf load_b_pm(const unsigned* w, int Npad, int kb, int cb, int lane) {
  FragU f;
  int col = cb + (lane & 15);
  int kd  = (kb + ((lane >> 4) << 3)) >> 1;
#pragma unroll
  for (int j = 0; j < 4; ++j) {
    f.u[j]     = w[(kd + j)     * Npad + col];
    f.u[4 + j] = w[(kd + 8 + j) * Npad + col];
  }
  return f.v;
}

DEVFN v8f wmma_bf16(v16bf a, v16bf b, v8f c) {
  return __builtin_amdgcn_wmma_f32_16x16x32_bf16(false, a, false, b, (short)0, c,
                                                 false, false);
}

// Async global -> LDS copy of 16 bytes (CDNA5 TDM-lite path, tracked by
// ASYNCcnt).  ldsAddr = LDS byte offset, gaddr = 64-bit global address.
DEVFN void async_g2l_b128(unsigned ldsAddr, unsigned long long gaddr) {
  asm volatile("global_load_async_to_lds_b128 %0, %1, off"
               :: "v"(ldsAddr), "v"(gaddr) : "memory");
}
DEVFN void wait_asynccnt0() {
  asm volatile("s_wait_asynccnt 0x0" ::: "memory");
}

// ---------------------------------------------------------------------------
// K0 kernels: prep
// ---------------------------------------------------------------------------
__global__ void k_cvt_bf16(const float* __restrict__ in, unsigned short* __restrict__ out, int n) {
  int i = blockIdx.x * blockDim.x + threadIdx.x;
  if (i < n) out[i] = f2bf(in[i]);
}

__global__ void k_pack_pair(const float* __restrict__ w, unsigned* __restrict__ o,
                            int K, int Npad, int Nsrc) {
  int d = blockIdx.x * blockDim.x + threadIdx.x;
  int total = (K >> 1) * Npad;
  if (d >= total) return;
  int n = d % Npad;
  int k = (d / Npad) << 1;
  unsigned lo = (n < Nsrc) ? (unsigned)f2bf(w[(size_t)k * Nsrc + n])       : 0u;
  unsigned hi = (n < Nsrc) ? (unsigned)f2bf(w[(size_t)(k + 1) * Nsrc + n]) : 0u;
  o[d] = lo | (hi << 16);
}

// scale = gamma*rsqrt(var+eps); shift = beta + (bias - mean)*scale
__global__ void k_fold(const float* __restrict__ bn, const float* __restrict__ bias,
                       float* __restrict__ so, float* __restrict__ to, int C) {
  int c = blockIdx.x * blockDim.x + threadIdx.x;
  if (c >= C) return;
  float g  = bn[0 * C + c], be = bn[1 * C + c];
  float m  = bn[2 * C + c], v  = bn[3 * C + c];
  float s  = g * rsqrtf(v + 1e-5f);
  so[c] = s;
  to[c] = be + (bias[c] - m) * s;
}

// ---------------------------------------------------------------------------
// K1: q/k/v GEMM.  grid = (ROWS/64, 3).  256 threads = 8 waves.
// wave w: rows r0 + (w>>1)*16, cols (w&1)*128 .. +127 (8 WMMA tiles).
// ---------------------------------------------------------------------------
__global__ void __launch_bounds__(256)
k_qkv(const unsigned* __restrict__ featsb,
      const unsigned* __restrict__ wq_p, const unsigned* __restrict__ wk_p,
      const unsigned* __restrict__ wv_p,
      const float* __restrict__ fold, const float* __restrict__ bv,
      float* __restrict__ qbuf, float* __restrict__ kbuf, float* __restrict__ vbuf) {
  int which = blockIdx.y;
  int r0    = blockIdx.x * 64;
  int wave  = threadIdx.x >> 5, lane = threadIdx.x & 31;
  int rowBase = r0 + (wave >> 1) * 16;
  int colBase = (wave & 1) * 128;
  const unsigned* wp = (which == 0) ? wq_p : (which == 1) ? wk_p : wv_p;

  v8f zero = {};
  v8f acc[8];
#pragma unroll
  for (int i = 0; i < 8; ++i) acc[i] = zero;

  int aRow = rowBase + (lane & 15);
  for (int kb = 0; kb < Cn; kb += 32) {
    v16bf a = load_a_dw(featsb, aRow, Cn / 2, kb, lane);
#pragma unroll
    for (int t = 0; t < 8; ++t) {
      v16bf b = load_b_pm(wp, Cn, kb, colBase + t * 16, lane);
      acc[t] = wmma_bf16(a, b, acc[t]);
    }
  }

  const float* sArr = fold;  const float* tArr = fold + 256;
  float* obuf = qbuf;
  if (which == 1) { sArr = fold + 512; tArr = fold + 768; obuf = kbuf; }
  else if (which == 2) { obuf = vbuf; }

#pragma unroll
  for (int t = 0; t < 8; ++t) {
    int col = colBase + t * 16 + (lane & 15);
#pragma unroll
    for (int i = 0; i < 8; ++i) {
      int row = rowBase + ((lane >> 4) << 3) + i;
      float x = acc[t][i];
      if (which == 2) x += bv[col];
      else            x = fmaxf(0.f, x * sArr[col] + tArr[col]);
      obuf[(size_t)row * Cn + col] = x;
    }
  }
}

// ---------------------------------------------------------------------------
// K2: fused per-point attention.  grid = ROWS blocks x 256 threads (8 waves).
// ---------------------------------------------------------------------------
__global__ void __launch_bounds__(256)
k_attn(const float* __restrict__ coords, const int* __restrict__ index,
       const float* __restrict__ qbuf, const float* __restrict__ kbuf,
       const float* __restrict__ vbuf,
       const float* __restrict__ pm_w1, const float* __restrict__ pm_b2,
       const float* __restrict__ pb_w1, const float* __restrict__ pb_b2,
       const unsigned* __restrict__ pmw2_p, const unsigned* __restrict__ pbw2_p,
       const unsigned* __restrict__ wew1_p,
       const float* __restrict__ fold,
       const float* __restrict__ we_w2, const float* __restrict__ we_b2,
       float* __restrict__ out) {
  __shared__ float          q_s[Cn];
  __shared__ float          kg[Sn][Cn];
  __shared__ float          vg[Sn][Cn];
  __shared__ float          pos_s[Sn][4];
  __shared__ unsigned short hh[Sn * Cn];      // MLP hidden (bf16); aliased as `rel`
  __shared__ float          pem[Sn][Cn];
  __shared__ float          att[Sn][16];
  __shared__ float          wsm[Sn][Gn];
  __shared__ int            nidx[Sn];

  int p = blockIdx.x;                 // flat (b, n)
  int b = p >> 12;                    // N = 4096
  int t = threadIdx.x, lane = t & 31, wave = t >> 5;

  if (t < Sn) nidx[t] = index[(size_t)p * Sn + t];
  __syncthreads();

  // ---- gather neighbors into LDS via async global->LDS (ASYNCcnt) ----
  {
    int s = t >> 4, c0 = (t & 15) << 4;
    int grow = (b << 12) + nidx[s];
    const float* krow = kbuf + (size_t)grow * Cn + c0;
    const float* vrow = vbuf + (size_t)grow * Cn + c0;
    unsigned ldsK = (unsigned)(size_t)&kg[s][c0];
    unsigned ldsV = (unsigned)(size_t)&vg[s][c0];
#pragma unroll
    for (int j = 0; j < 4; ++j) {
      async_g2l_b128(ldsK + 16u * j, (unsigned long long)(krow + 4 * j));
      async_g2l_b128(ldsV + 16u * j, (unsigned long long)(vrow + 4 * j));
    }
    q_s[t] = qbuf[(size_t)p * Cn + t];
    if (t < Sn) {
      int gr = (b << 12) + nidx[t];
#pragma unroll
      for (int d = 0; d < 3; ++d)
        pos_s[t][d] = coords[(size_t)p * 3 + d] - coords[(size_t)gr * 3 + d];
    }
    wait_asynccnt0();
  }
  __syncthreads();

  // ---- pm hidden: h = relu((pos @ pm_w1) * s_pm + t_pm)  (K=3, VALU) ----
  {
    const float* s_pm = fold + 1024; const float* t_pm = fold + 1280;
    for (int e = t; e < Sn * Cn; e += 256) {
      int s = e >> 8, c = e & 255;
      float d = pos_s[s][0] * pm_w1[c] + pos_s[s][1] * pm_w1[Cn + c] +
                pos_s[s][2] * pm_w1[2 * Cn + c];
      hh[e] = f2bf(fmaxf(0.f, d * s_pm[c] + t_pm[c]));
    }
  }
  __syncthreads();

  // ---- pem = h @ pm_w2 + pm_b2  (WMMA, wave w owns cols w*32..w*32+31) ----
  const unsigned* hdw = (const unsigned*)hh;
  int cb = wave * 32;
  {
    v8f a0 = {}, a1 = {};
    for (int kb = 0; kb < Cn; kb += 32) {
      v16bf a  = load_a_dw(hdw, lane & 15, Cn / 2, kb, lane);
      v16bf b0 = load_b_pm(pmw2_p, Cn, kb, cb, lane);
      v16bf b1 = load_b_pm(pmw2_p, Cn, kb, cb + 16, lane);
      a0 = wmma_bf16(a, b0, a0);
      a1 = wmma_bf16(a, b1, a1);
    }
#pragma unroll
    for (int i = 0; i < 8; ++i) {
      int row = ((lane >> 4) << 3) + i;
      int c0  = cb + (lane & 15);
      int c1  = c0 + 16;
      pem[row][c0] = a0[i] + pm_b2[c0];
      pem[row][c1] = a1[i] + pm_b2[c1];
    }
  }
  __syncthreads();   // pem done; hh free for reuse

  // ---- pb hidden ----
  {
    const float* s_pb = fold + 1536; const float* t_pb = fold + 1792;
    for (int e = t; e < Sn * Cn; e += 256) {
      int s = e >> 8, c = e & 255;
      float d = pos_s[s][0] * pb_w1[c] + pos_s[s][1] * pb_w1[Cn + c] +
                pos_s[s][2] * pb_w1[2 * Cn + c];
      hh[e] = f2bf(fmaxf(0.f, d * s_pb[c] + t_pb[c]));
    }
  }
  __syncthreads();

  // ---- peb GEMM; epilogue fuses relation_qk and vg update (peb never stored) ----
  {
    v8f a0 = {}, a1 = {};
    for (int kb = 0; kb < Cn; kb += 32) {
      v16bf a  = load_a_dw(hdw, lane & 15, Cn / 2, kb, lane);
      v16bf b0 = load_b_pm(pbw2_p, Cn, kb, cb, lane);
      v16bf b1 = load_b_pm(pbw2_p, Cn, kb, cb + 16, lane);
      a0 = wmma_bf16(a, b0, a0);
      a1 = wmma_bf16(a, b1, a1);
    }
    __syncthreads();  // all waves done reading hh -> safe to alias rel over hh
    unsigned short* rel = hh;
#pragma unroll
    for (int i = 0; i < 8; ++i) {
      int row = ((lane >> 4) << 3) + i;
#pragma unroll
      for (int half = 0; half < 2; ++half) {
        int c   = cb + half * 16 + (lane & 15);
        float pe = (half ? a1[i] : a0[i]) + pb_b2[c];
        float r  = (kg[row][c] - q_s[c]) * pem[row][c] + pe;
        rel[row * Cn + c] = f2bf(r);
        vg[row][c] += pe;
      }
    }
  }
  __syncthreads();

  // ---- attention logits: relu(bn(rel @ we_w1)) on wave 0 (N padded to 16) ----
  if (wave == 0) {
    v8f a0 = {};
    const unsigned* rdw = (const unsigned*)hh;   // == rel
    for (int kb = 0; kb < Cn; kb += 32) {
      v16bf a = load_a_dw(rdw, lane & 15, Cn / 2, kb, lane);
      v16bf b = load_b_pm(wew1_p, 16, kb, 0, lane);
      a0 = wmma_bf16(a, b, a0);
    }
    const float* s_we = fold + 2048; const float* t_we = fold + 2056;
    int col = lane & 15;
#pragma unroll
    for (int i = 0; i < 8; ++i) {
      int row = ((lane >> 4) << 3) + i;
      float x = 0.f;
      if (col < Gn) x = fmaxf(0.f, a0[i] * s_we[col] + t_we[col]);
      att[row][col] = x;
    }
  }
  __syncthreads();

  // ---- tiny 8x8 GEMM + softmax over S (one thread per output group) ----
  if (t < Gn) {
    float lg[Sn], mx = -3.4e38f;
#pragma unroll
    for (int s = 0; s < Sn; ++s) {
      float l = we_b2[t];
#pragma unroll
      for (int g = 0; g < Gn; ++g) l += att[s][g] * we_w2[g * Gn + t];
      lg[s] = l; mx = fmaxf(mx, l);
    }
    float sum = 0.f;
#pragma unroll
    for (int s = 0; s < Sn; ++s) { lg[s] = __expf(lg[s] - mx); sum += lg[s]; }
    float inv = 1.f / sum;
#pragma unroll
    for (int s = 0; s < Sn; ++s) wsm[s][t] = lg[s] * inv;
  }
  __syncthreads();

  // ---- out[c] = sum_s vg[s][c] * w[s][c/32] ----
  {
    int c = t, g = c >> 5;
    float o = 0.f;
#pragma unroll
    for (int s = 0; s < Sn; ++s) o += vg[s][c] * wsm[s][g];
    out[(size_t)p * Cn + c] = o;
  }
}

// ---------------------------------------------------------------------------
extern "C" void kernel_launch(void* const* d_in, const int* in_sizes, int n_in,
                              void* d_out, int out_size, void* d_ws, size_t ws_size,
                              hipStream_t stream) {
  (void)in_sizes; (void)n_in; (void)out_size; (void)ws_size;
  const float* feats  = (const float*)d_in[0];
  const float* coords = (const float*)d_in[1];
  const int*   index  = (const int*)d_in[2];
  const float* wq     = (const float*)d_in[3];
  const float* bq     = (const float*)d_in[4];
  const float* bnq    = (const float*)d_in[5];
  const float* wk     = (const float*)d_in[6];
  const float* bk     = (const float*)d_in[7];
  const float* bnk    = (const float*)d_in[8];
  const float* wv     = (const float*)d_in[9];
  const float* bv     = (const float*)d_in[10];
  const float* pm_w1  = (const float*)d_in[11];
  const float* pm_b1  = (const float*)d_in[12];
  const float* pm_bn  = (const float*)d_in[13];
  const float* pm_w2  = (const float*)d_in[14];
  const float* pm_b2  = (const float*)d_in[15];
  const float* pb_w1  = (const float*)d_in[16];
  const float* pb_b1  = (const float*)d_in[17];
  const float* pb_bn  = (const float*)d_in[18];
  const float* pb_w2  = (const float*)d_in[19];
  const float* pb_b2  = (const float*)d_in[20];
  const float* we_w1  = (const float*)d_in[21];
  const float* we_b1  = (const float*)d_in[22];
  const float* we_bn  = (const float*)d_in[23];
  const float* we_w2  = (const float*)d_in[24];
  const float* we_b2  = (const float*)d_in[25];

  char* ws = (char*)d_ws;
  float*          qbuf   = (float*)(ws + OFF_Q);
  float*          kbuf   = (float*)(ws + OFF_K);
  float*          vbuf   = (float*)(ws + OFF_V);
  unsigned short* featsb = (unsigned short*)(ws + OFF_FEATSB);
  unsigned*       wq_p   = (unsigned*)(ws + OFF_WQ);
  unsigned*       wk_p   = (unsigned*)(ws + OFF_WK);
  unsigned*       wv_p   = (unsigned*)(ws + OFF_WV);
  unsigned*       pmw2_p = (unsigned*)(ws + OFF_PMW2);
  unsigned*       pbw2_p = (unsigned*)(ws + OFF_PBW2);
  unsigned*       wew1_p = (unsigned*)(ws + OFF_WEW1);
  float*          fold   = (float*)(ws + OFF_FOLD);

  // ---- K0: prep ----
  int nfeat = ROWS * Cn;
  k_cvt_bf16<<<(nfeat + 255) / 256, 256, 0, stream>>>(feats, featsb, nfeat);
  int sqDw = (Cn / 2) * Cn;   // 32768 dwords per square weight
  k_pack_pair<<<(sqDw + 255) / 256, 256, 0, stream>>>(wq,    wq_p,   Cn, Cn, Cn);
  k_pack_pair<<<(sqDw + 255) / 256, 256, 0, stream>>>(wk,    wk_p,   Cn, Cn, Cn);
  k_pack_pair<<<(sqDw + 255) / 256, 256, 0, stream>>>(wv,    wv_p,   Cn, Cn, Cn);
  k_pack_pair<<<(sqDw + 255) / 256, 256, 0, stream>>>(pm_w2, pmw2_p, Cn, Cn, Cn);
  k_pack_pair<<<(sqDw + 255) / 256, 256, 0, stream>>>(pb_w2, pbw2_p, Cn, Cn, Cn);
  k_pack_pair<<<((Cn / 2) * 16 + 255) / 256, 256, 0, stream>>>(we_w1, wew1_p, Cn, 16, Gn);
  k_fold<<<1, 256, 0, stream>>>(bnq,   bq,    fold + 0,    fold + 256,  Cn);
  k_fold<<<1, 256, 0, stream>>>(bnk,   bk,    fold + 512,  fold + 768,  Cn);
  k_fold<<<1, 256, 0, stream>>>(pm_bn, pm_b1, fold + 1024, fold + 1280, Cn);
  k_fold<<<1, 256, 0, stream>>>(pb_bn, pb_b1, fold + 1536, fold + 1792, Cn);
  k_fold<<<1, 256, 0, stream>>>(we_bn, we_b1, fold + 2048, fold + 2056, Gn);

  // ---- K1: q/k/v ----
  k_qkv<<<dim3(ROWS / 64, 3), 256, 0, stream>>>(
      (const unsigned*)featsb, wq_p, wk_p, wv_p, fold, bv, qbuf, kbuf, vbuf);

  // ---- K2: fused attention ----
  k_attn<<<ROWS, 256, 0, stream>>>(
      coords, index, qbuf, kbuf, vbuf,
      pm_w1, pm_b2, pb_w1, pb_b2,
      pmw2_p, pbw2_p, wew1_p,
      fold, we_w2, we_b2, (float*)d_out);
}
